// SimpleVSN_47467978555513
// MI455X (gfx1250) — matrix operand order repaired and checked by reading, not compiled, not run on hardware
//
#include <hip/hip_runtime.h>

// ---------------------------------------------------------------------------
// Fused SimpleVSN for MI455X (gfx1250, wave32, WMMA).
// B=64 T=512 F=32 H=64.  ~13 GFLOP vs ~15MB unique traffic -> matrix-bound.
// Strategy: one 8-wave workgroup per 16-token tile; all GEMMs use
// v_wmma_f32_16x16x32_f16; v[16 x F*H] lives only in LDS (never in HBM).
// ---------------------------------------------------------------------------

typedef __attribute__((ext_vector_type(16))) _Float16 v16h;
typedef __attribute__((ext_vector_type(8)))  _Float16 v8h;
typedef __attribute__((ext_vector_type(4)))  _Float16 v4h;
typedef __attribute__((ext_vector_type(8)))  float    v8f;

#define Bsz  64
#define Tsz  512
#define Fsz  32
#define Hsz  64
#define BT   (Bsz * Tsz)
#define VPAD 72   // padded H stride (halves): 144B rows -> 16B-aligned, 16 distinct banks

// A-matrix (16x32 f16) lane/element -> K map from ISA 7.12.2:
// lane half hs = lane>>4;  j<8: K = hs*8 + j ;  j>=8: K = 16 + hs*8 + (j-8)
__device__ __forceinline__ int kmapA(int j, int hs) {
  return (j < 8) ? (hs * 8 + j) : (16 + hs * 8 + (j - 8));
}
// B-matrix (32x16 f16) packed with the same K map as A: K = kmapA(j, hs), N = lane&15

// --- prep: pack W2 [F,H,H] f32 -> f16 B-fragments --------------------------
// layout: idx = (((f*2 + c)*4 + nt)*32 + lane)*16 + j  -> contiguous v16h/lane
__global__ void pack_w2(const float* __restrict__ W2, _Float16* __restrict__ W2h) {
  int idx = blockIdx.x * blockDim.x + threadIdx.x;
  if (idx >= Fsz * 2 * 4 * 32 * 16) return;
  int j    = idx & 15;
  int lane = (idx >> 4) & 31;
  int nt   = (idx >> 9) & 3;
  int c    = (idx >> 11) & 1;
  int f    = idx >> 12;
  int hs = lane >> 4, lm = lane & 15;
  int k = c * 32 + kmapA(j, hs);     // K (the h index)
  int n = nt * 16 + lm;              // N (the output k index)
  W2h[idx] = (_Float16)W2[(f * Hsz + k) * Hsz + n];
}

// --- prep: pack Wsel [F, F*H] f32 -> f16 B-fragments -----------------------
// layout: idx = ((kc*2 + nt)*32 + lane)*16 + j
__global__ void pack_wsel(const float* __restrict__ Wsel, _Float16* __restrict__ Wh) {
  int idx = blockIdx.x * blockDim.x + threadIdx.x;
  if (idx >= 64 * 2 * 32 * 16) return;
  int j    = idx & 15;
  int lane = (idx >> 4) & 31;
  int nt   = (idx >> 9) & 1;
  int kc   = idx >> 10;
  int hs = lane >> 4, lm = lane & 15;
  int d    = kc * 32 + kmapA(j, hs); // K over flattened F*H
  int fcol = nt * 16 + lm;           // N = selector feature
  Wh[idx] = (_Float16)Wsel[fcol * (Fsz * Hsz) + d];
}

// --- fused main kernel -----------------------------------------------------
__global__ void __launch_bounds__(256) vsn_fused(
    const float* __restrict__ x,   const float* __restrict__ W1,
    const float* __restrict__ b1,  const float* __restrict__ b2,
    const float* __restrict__ bsel,
    const _Float16* __restrict__ W2h, const _Float16* __restrict__ Wselh,
    float* __restrict__ out)
{
  extern __shared__ unsigned char smem[];
  _Float16* vlds = (_Float16*)smem;                         // [F][16][VPAD] f16
  float*    xs   = (float*)(smem + Fsz * 16 * VPAD * 2);    // [16][F]
  float*    lg   = xs + 16 * Fsz;                           // [16][F] logits->weights

  const int tid  = threadIdx.x;
  const int wave = tid >> 5;
  const int lane = tid & 31;
  const int hs   = lane >> 4;
  const int lm   = lane & 15;
  const int t0   = blockIdx.x * 16;

  // cache-warm the packed weights this wave will consume (global_prefetch_b8)
  __builtin_prefetch(W2h + ((size_t)(wave * 4) * 2 * 4 * 32 * 16), 0, 3);
  __builtin_prefetch(Wselh + ((size_t)(wave * 8) * 2 * 32 * 16), 0, 3);

  // Phase A: stage x tile, seed logits with bsel
  {
    const float2 xv = ((const float2*)(x + (size_t)t0 * Fsz))[tid];
    ((float2*)xs)[tid] = xv;
    int e0 = tid * 2;
    lg[e0]     = bsel[e0 & 31];
    lg[e0 + 1] = bsel[(e0 + 1) & 31];
  }
  __syncthreads();

  // Phase B: per-feature v = relu(x*W1+b1) @ W2 + b2 ; each wave owns 4 features
  for (int fi = 0; fi < 4; ++fi) {
    const int f = wave * 4 + fi;
    const float xm = xs[lm * Fsz + f];        // x[token m=lm][f]

    v8f acc[4];
    #pragma unroll
    for (int nt = 0; nt < 4; ++nt) {
      const float bv = b2[f * Hsz + nt * 16 + lm];   // C init = b2 broadcast over M
      #pragma unroll
      for (int r = 0; r < 8; ++r) acc[nt][r] = bv;
    }

    #pragma unroll
    for (int c = 0; c < 2; ++c) {
      // h A-fragment: K groups are contiguous in W1/b1 -> 16B-aligned b128 loads
      const float* w1r = W1 + f * Hsz;
      const float* b1r = b1 + f * Hsz;
      float wlo[8], blo[8], whi[8], bhi[8];
      *(float4*)&wlo[0] = *(const float4*)(w1r + c * 32 + hs * 8);
      *(float4*)&wlo[4] = *(const float4*)(w1r + c * 32 + hs * 8 + 4);
      *(float4*)&whi[0] = *(const float4*)(w1r + c * 32 + 16 + hs * 8);
      *(float4*)&whi[4] = *(const float4*)(w1r + c * 32 + 16 + hs * 8 + 4);
      *(float4*)&blo[0] = *(const float4*)(b1r + c * 32 + hs * 8);
      *(float4*)&blo[4] = *(const float4*)(b1r + c * 32 + hs * 8 + 4);
      *(float4*)&bhi[0] = *(const float4*)(b1r + c * 32 + 16 + hs * 8);
      *(float4*)&bhi[4] = *(const float4*)(b1r + c * 32 + 16 + hs * 8 + 4);

      v16h a;
      #pragma unroll
      for (int j = 0; j < 8; ++j) {
        float h0 = xm * wlo[j] + blo[j];
        float h1 = xm * whi[j] + bhi[j];
        a[j]     = (_Float16)(h0 > 0.f ? h0 : 0.f);
        a[j + 8] = (_Float16)(h1 > 0.f ? h1 : 0.f);
      }

      #pragma unroll
      for (int nt = 0; nt < 4; ++nt) {
        const v16h bfrag =
            *(const v16h*)(W2h + ((((f * 2 + c) * 4 + nt) * 32 + lane) << 4));
        acc[nt] = __builtin_amdgcn_wmma_f32_16x16x32_f16(
            false, a, false, bfrag, (short)0, acc[nt], false, false);
      }
    }

    // spill v tile to LDS as f16 (D layout: lane=col n, VGPR r -> row r+8*hs)
    #pragma unroll
    for (int nt = 0; nt < 4; ++nt)
      #pragma unroll
      for (int r = 0; r < 8; ++r) {
        const int m = r + 8 * hs;
        vlds[(f * 16 + m) * VPAD + nt * 16 + lm] = (_Float16)acc[nt][r];
      }
  }
  __syncthreads();

  // Phase C: logits += flat(v) @ Wsel^T ; each wave covers its 8 K-chunks
  {
    v8f dacc[2];
    #pragma unroll
    for (int nt = 0; nt < 2; ++nt)
      #pragma unroll
      for (int r = 0; r < 8; ++r) dacc[nt][r] = 0.f;

    for (int fi = 0; fi < 4; ++fi) {
      const int f = wave * 4 + fi;
      #pragma unroll
      for (int c = 0; c < 2; ++c) {
        const int kc = f * 2 + c;
        // A-fragment straight from LDS: two 16B-aligned ds_load_b128
        const _Float16* row = vlds + (f * 16 + lm) * VPAD + c * 32 + hs * 8;
        const v8h lo = *(const v8h*)(row);
        const v8h hi = *(const v8h*)(row + 16);
        v16h a;
        #pragma unroll
        for (int j = 0; j < 8; ++j) { a[j] = lo[j]; a[j + 8] = hi[j]; }

        #pragma unroll
        for (int nt = 0; nt < 2; ++nt) {
          const v16h bfrag =
              *(const v16h*)(Wselh + (((kc * 2 + nt) * 32 + lane) << 4));
          dacc[nt] = __builtin_amdgcn_wmma_f32_16x16x32_f16(
              false, a, false, bfrag, (short)0, dacc[nt], false, false);
        }
      }
    }
    // cross-wave K reduction via LDS float atomics (ds_add_f32)
    #pragma unroll
    for (int nt = 0; nt < 2; ++nt)
      #pragma unroll
      for (int r = 0; r < 8; ++r) {
        const int m = r + 8 * hs;
        atomicAdd(&lg[m * Fsz + nt * 16 + lm], dacc[nt][r]);
      }
  }
  __syncthreads();

  // Phase D: softmax over F (one thread per token, two-pass, in-place)
  if (tid < 16) {
    const int m = tid;
    float mx = lg[m * Fsz];
    for (int f2 = 1; f2 < Fsz; ++f2) mx = fmaxf(mx, lg[m * Fsz + f2]);
    float s = 0.f;
    for (int f2 = 0; f2 < Fsz; ++f2) s += __expf(lg[m * Fsz + f2] - mx);
    const float inv = 1.f / s;
    for (int f2 = 0; f2 < Fsz; ++f2)
      lg[m * Fsz + f2] = __expf(lg[m * Fsz + f2] - mx) * inv;
  }
  __syncthreads();

  // Phase E: out[m, k0..k0+3] = sum_f w[m,f] * v[m,f,k]  (4 outputs/thread)
  {
    const int m  = tid >> 4;
    const int k0 = (tid & 15) * 4;
    float a0 = 0.f, a1 = 0.f, a2 = 0.f, a3 = 0.f;
    for (int f2 = 0; f2 < Fsz; ++f2) {
      const float w = lg[m * Fsz + f2];
      const v4h vv = *(const v4h*)(vlds + (f2 * 16 + m) * VPAD + k0);  // 8B-aligned
      a0 += w * (float)vv[0];
      a1 += w * (float)vv[1];
      a2 += w * (float)vv[2];
      a3 += w * (float)vv[3];
    }
    float4 o; o.x = a0; o.y = a1; o.z = a2; o.w = a3;
    *(float4*)(out + ((size_t)(t0 + m)) * Hsz + k0) = o;   // global_store_b128
  }
}

// ---------------------------------------------------------------------------
extern "C" void kernel_launch(void* const* d_in, const int* in_sizes, int n_in,
                              void* d_out, int out_size, void* d_ws, size_t ws_size,
                              hipStream_t stream) {
  (void)in_sizes; (void)n_in; (void)out_size; (void)ws_size;
  const float* x    = (const float*)d_in[0];
  const float* W1   = (const float*)d_in[1];
  const float* b1   = (const float*)d_in[2];
  const float* W2   = (const float*)d_in[3];
  const float* b2   = (const float*)d_in[4];
  const float* Wsel = (const float*)d_in[5];
  const float* bsel = (const float*)d_in[6];
  float* out = (float*)d_out;

  // workspace: packed f16 weight fragments (256KB + 128KB)
  _Float16* W2h   = (_Float16*)d_ws;
  _Float16* Wselh = W2h + (Fsz * 2 * 4 * 32 * 16);   // 131072 halves

  pack_w2  <<<(Fsz * 2 * 4 * 32 * 16) / 256, 256, 0, stream>>>(W2, W2h);
  pack_wsel<<<(64 * 2 * 32 * 16)      / 256, 256, 0, stream>>>(Wsel, Wselh);

  const size_t smem = (size_t)Fsz * 16 * VPAD * 2   // v tile (f16)
                    + (size_t)16 * Fsz * 4          // x tile
                    + (size_t)16 * Fsz * 4;         // logits/weights
  // 76KB dynamic LDS: fine for gfx1250's 320KB WGP LDS; opt in explicitly.
  (void)hipFuncSetAttribute((const void*)vsn_fused,
                            hipFuncAttributeMaxDynamicSharedMemorySize,
                            (int)smem);
  vsn_fused<<<BT / 16, 256, smem, stream>>>(x, W1, b1, b2, bsel, W2h, Wselh, out);
}
